// VGG_40063454937478
// MI455X (gfx1250) — matrix operand order repaired and checked by reading, not compiled, not run on hardware
//
#include <hip/hip_runtime.h>
#include <hip/hip_bf16.h>

// ---------------------------------------------------------------------------
// MI455X (gfx1250) implementation.
// Convs = implicit GEMM on v_wmma_f32_16x16x32_bf16 (bf16 A/B, f32 accum),
//         2x2 register blocking per wave (4 WMMAs / K-step / wave),
//         async global->LDS staging of the weight tile when available.
// _zzz  = wave-per-sample gradient estimation + RMSprop on per-pixel (A,B).
// ---------------------------------------------------------------------------

typedef unsigned short bf16_t;
typedef unsigned int   u32;

typedef __attribute__((ext_vector_type(16))) __bf16        v16bf;
typedef __attribute__((ext_vector_type(8)))  float         v8f;
typedef __attribute__((ext_vector_type(8)))  unsigned int  v8u;

#if defined(__has_builtin)
#if __has_builtin(__builtin_amdgcn_global_load_async_to_lds_b128) && \
    __has_builtin(__builtin_amdgcn_s_wait_asynccnt)
#define HAVE_ASYNC_LDS 1
#endif
#endif

#ifdef HAVE_ASYNC_LDS
// toolchain models b128 async copies as pointers to gcc-vector int4
typedef int v4i_g __attribute__((vector_size(16)));
typedef __attribute__((address_space(1))) v4i_g gv4i;   // global
typedef __attribute__((address_space(3))) v4i_g lv4i;   // LDS
#endif

__device__ __forceinline__ bf16_t f2bf(float f) {
    u32 u = __builtin_bit_cast(u32, f);
    u += 0x7FFFu + ((u >> 16) & 1u);          // round-to-nearest-even
    return (bf16_t)(u >> 16);
}
__device__ __forceinline__ float bf2f(bf16_t h) {
    u32 u = ((u32)h) << 16;
    return __builtin_bit_cast(float, u);
}
__device__ __forceinline__ u32 hash3(u32 a, u32 b, u32 c) {
    u32 h = a * 0x9E3779B1u;
    h ^= b + 0x85EBCA77u + (h << 6) + (h >> 2);
    h *= 0xC2B2AE3Du;
    h ^= c + 0x27D4EB2Fu + (h << 6) + (h >> 2);
    h ^= h >> 16; h *= 0x7FEB352Du; h ^= h >> 15; h *= 0x846CA68Bu; h ^= h >> 16;
    return h;
}

// ---------------------------------------------------------------------------
// Elementwise helpers
// ---------------------------------------------------------------------------
__global__ void k_f32_to_bf16(const float* __restrict__ in, bf16_t* __restrict__ out, int total) {
    int i = blockIdx.x * blockDim.x + threadIdx.x;
    if (i < total) out[i] = f2bf(in[i]);
}

// repack conv weight (Cout,Cin,3,3) fp32 -> bf16 [Cout, Kpad], zero-padded K
__global__ void k_pack_weights(const float* __restrict__ w, bf16_t* __restrict__ out,
                               int Cout, int K, int Kpad) {
    int i = blockIdx.x * blockDim.x + threadIdx.x;
    if (i >= Cout * Kpad) return;
    int co = i / Kpad, k = i - co * Kpad;
    out[i] = (k < K) ? f2bf(w[(size_t)co * K + k]) : (bf16_t)0;
}

// per-channel BN fold: scale = g*rsqrt(v+eps), shift = scale*(bias? - m) + b
__global__ void k_scale_shift(const float* __restrict__ g, const float* __restrict__ bb,
                              const float* __restrict__ m, const float* __restrict__ v,
                              const float* __restrict__ bias,
                              float* __restrict__ scale, float* __restrict__ shift,
                              int C, int fold_bias) {
    int c = blockIdx.x * blockDim.x + threadIdx.x;
    if (c >= C) return;
    float inv = g[c] * rsqrtf(v[c] + 1e-5f);
    float bs  = fold_bias ? bias[c] : 0.0f;
    scale[c] = inv;
    shift[c] = inv * (bs - m[c]) + bb[c];
}

// ---------------------------------------------------------------------------
// Implicit-GEMM conv3x3 (pad=1) on WMMA bf16.
//   out[Cout, B*H*W] = W[Cout, Cin*9] x im2col(act)[Cin*9, B*H*W]
// Block: 256 threads = 8 waves. Block tile 64(M) x 128(N), K stepped by 32.
// Wave (wm = wave&1, wn = wave>>1) owns a 32x32 tile: 4 accumulators,
// 4 v_wmma_f32_16x16x32_bf16 per K-step.
// LDS: A-tile [64][32] stride 40 halfs (16B-aligned rows, spread banks)
//      B-tile [128][32] n-major stride 36 halfs (k-pairs contiguous -> b32 reads)
// mode 0: out = relu(scale*acc + shift)    (bias folded into shift)
// mode 1: out = acc + bias                 (x1 for _zzz layers)
// ---------------------------------------------------------------------------
#define A_STRIDE 40
#define B_STRIDE 36

__global__ __launch_bounds__(256) void k_conv_wmma(
    const bf16_t* __restrict__ act, const bf16_t* __restrict__ wmat,
    const float* __restrict__ scale, const float* __restrict__ shift,
    const float* __restrict__ bias, bf16_t* __restrict__ out,
    int Cin, int Cout, int H, int W, int Kact, int Kpad, int mode) {

    __shared__ bf16_t As[64 * A_STRIDE];    // weights tile   [m][k]
    __shared__ bf16_t Bs[128 * B_STRIDE];   // im2col tile    [n][k]

    const int tid  = threadIdx.x;
    const int lane = tid & 31;
    const int wave = tid >> 5;
    const int wm   = wave & 1;       // 2 M sub-tiles of 32
    const int wn   = wave >> 1;      // 4 N sub-tiles of 32
    const int HW   = H * W;
    const int m0   = blockIdx.y * 64;
    const int n0   = blockIdx.x * 128;

    // A-tile cooperative staging coords (16B per thread, covers 64x32 exactly)
    const int arow = tid >> 2;
    const int akc  = (tid & 3) * 8;
    const bf16_t* awsrc = wmat + (size_t)(m0 + arow) * Kpad + akc;
    bf16_t* aldst = &As[arow * A_STRIDE + akc];

    v8f acc00 = {}, acc01 = {}, acc10 = {}, acc11 = {};

    for (int k0 = 0; k0 < Kpad; k0 += 32) {
        // --- stage A (weights) 64x32 bf16: async DMA to LDS when available
#ifdef HAVE_ASYNC_LDS
        __builtin_amdgcn_global_load_async_to_lds_b128(
            (gv4i*)(awsrc + k0), (lv4i*)aldst, 0, 0);
#else
        *(uint4*)aldst = *(const uint4*)(awsrc + k0);
#endif
        __builtin_prefetch(awsrc + k0 + 32, 0, 0);   // global_prefetch next chunk

        // --- stage B (im2col gather): 4 chunks of 4 k-halfs, n-major layout
        #pragma unroll
        for (int q = 0; q < 4; ++q) {
            int e   = q * 256 + tid;       // 0..1023 chunks
            int nl  = e >> 3;              // 0..127
            int kc  = (e & 7) * 4;         // 0..28
            int n   = n0 + nl;
            int b   = n / HW, r = n - b * HW;
            int oh  = r / W, ow = r - oh * W;
            u32 p0 = 0, p1 = 0;
            #pragma unroll
            for (int j = 0; j < 4; ++j) {
                int kg = k0 + kc + j;
                bf16_t v = 0;
                if (kg < Kact) {
                    int ci  = kg / 9, rem = kg - ci * 9;
                    int kh  = rem / 3, kw = rem - kh * 3;
                    int ih  = oh + kh - 1, iw = ow + kw - 1;
                    if ((unsigned)ih < (unsigned)H && (unsigned)iw < (unsigned)W)
                        v = act[((size_t)(b * Cin + ci)) * HW + ih * W + iw];
                }
                if (j < 2) p0 |= ((u32)v) << (16 * j);
                else       p1 |= ((u32)v) << (16 * (j - 2));
            }
            uint2 pk; pk.x = p0; pk.y = p1;
            *(uint2*)&Bs[nl * B_STRIDE + kc] = pk;   // 8B aligned
        }
#ifdef HAVE_ASYNC_LDS
        __builtin_amdgcn_s_wait_asynccnt(0);
#endif
        __syncthreads();

        // --- build fragments per CDNA5 16-bit WMMA VGPR layout
        const int l15 = lane & 15;
        const int kq  = (lane >> 4) * 8;   // A halves
        const int kb  = (lane >> 4) * 16;  // unused: B reads use kq too
        v8u a0r, a1r, b0r, b1r;
        {
            int ml0 = wm * 32 + l15;
            int ml1 = ml0 + 16;
            #pragma unroll
            for (int j = 0; j < 4; ++j) {
                a0r[j]     = *(const u32*)&As[ml0 * A_STRIDE + kq + 2 * j];
                a0r[4 + j] = *(const u32*)&As[ml0 * A_STRIDE + 16 + kq + 2 * j];
                a1r[j]     = *(const u32*)&As[ml1 * A_STRIDE + kq + 2 * j];
                a1r[4 + j] = *(const u32*)&As[ml1 * A_STRIDE + 16 + kq + 2 * j];
            }
            int nn0 = wn * 32 + l15;
            int nn1 = nn0 + 16;
            #pragma unroll
            for (int j = 0; j < 8; ++j) {
                b0r[j] = *(const u32*)&Bs[nn0 * B_STRIDE + (j >> 2) * 16 + kq + 2 * (j & 3)];
                b1r[j] = *(const u32*)&Bs[nn1 * B_STRIDE + (j >> 2) * 16 + kq + 2 * (j & 3)];
            }
        }
        v16bf a0 = __builtin_bit_cast(v16bf, a0r);
        v16bf a1 = __builtin_bit_cast(v16bf, a1r);
        v16bf b0 = __builtin_bit_cast(v16bf, b0r);
        v16bf b1 = __builtin_bit_cast(v16bf, b1r);
        acc00 = __builtin_amdgcn_wmma_f32_16x16x32_bf16(false, a0, false, b0, (short)0, acc00, false, false);
        acc01 = __builtin_amdgcn_wmma_f32_16x16x32_bf16(false, a0, false, b1, (short)0, acc01, false, false);
        acc10 = __builtin_amdgcn_wmma_f32_16x16x32_bf16(false, a1, false, b0, (short)0, acc10, false, false);
        acc11 = __builtin_amdgcn_wmma_f32_16x16x32_bf16(false, a1, false, b1, (short)0, acc11, false, false);
        __syncthreads();
    }

    // --- epilogue: 4 tiles; lane holds column n, VGPR j holds row m
    const v8f* accs[4] = { &acc00, &acc01, &acc10, &acc11 };
    #pragma unroll
    for (int t = 0; t < 4; ++t) {
        int ti = t >> 1, tj = t & 1;     // ti: M sub-tile, tj: N sub-tile
        int n  = n0 + wn * 32 + tj * 16 + (lane & 15);
        int b  = n / HW, r = n - b * HW;
        int mb = m0 + wm * 32 + ti * 16 + (lane >> 4) * 8;
        const v8f a = *accs[t];
        #pragma unroll
        for (int j = 0; j < 8; ++j) {
            int m = mb + j;
            float v = a[j];
            if (mode == 0) {
                v = scale[m] * v + shift[m];
                v = v > 0.0f ? v : 0.0f;
            } else {
                v += bias[m];
            }
            out[((size_t)(b * Cout + m)) * HW + r] = f2bf(v);
        }
    }
}

// ---------------------------------------------------------------------------
// _zzz: per-pixel affine (A,B) fitted with 10 RMSprop steps
// ---------------------------------------------------------------------------
__global__ void k_zzz_init(float* A, float* B, float* vA, float* vB,
                           float* gA, float* gB, int total) {
    int i = blockIdx.x * blockDim.x + threadIdx.x;
    if (i >= total) return;
    A[i] = 1.0f; B[i] = 0.0f; vA[i] = 0.0f; vB[i] = 0.0f; gA[i] = 0.0f; gB[i] = 0.0f;
}

// One wave per (batch b, sample s): triple (i0,i1,i2), channel-strided lanes.
__global__ __launch_bounds__(256) void k_zzz_grad(
    const bf16_t* __restrict__ xin,   // layer input  [B,Cin,H,W]
    const bf16_t* __restrict__ x1,    // conv out + bias [B,C,H,W]
    const float* __restrict__ A, const float* __restrict__ B,
    float* __restrict__ gA, float* __restrict__ gB,
    int Cin, int C, int H, int W, int Bn, int sample_num, int step, int layer) {

    int gwarp = (blockIdx.x * blockDim.x + threadIdx.x) >> 5;
    int lane  = threadIdx.x & 31;
    if (gwarp >= Bn * sample_num) return;
    int b = gwarp / sample_num, s = gwarp - b * sample_num;
    int HW = H * W;

    u32 seed = (u32)(layer * 131 + step);
    int i0 = (int)(hash3(seed, (u32)b, (u32)(3 * s + 0)) % (u32)HW);
    int i1 = (int)(hash3(seed, (u32)b, (u32)(3 * s + 1)) % (u32)HW);
    int i2 = (int)(hash3(seed, (u32)b, (u32)(3 * s + 2)) % (u32)HW);

    size_t base = (size_t)b * HW;
    float A0 = A[base + i0], B0 = B[base + i0];
    float A1 = A[base + i1], B1 = B[base + i1];
    float A2 = A[base + i2], B2 = B[base + i2];

    // accumulate over conv-output channels
    float t1s = 0, t2s = 0, pa0 = 0, pb0 = 0, pa1 = 0, pb1 = 0, pa2 = 0, pb2 = 0;
    for (int c = lane; c < C; c += 32) {
        size_t cb = ((size_t)(b * C + c)) * HW;
        float v0 = bf2f(x1[cb + i0]);
        float v1 = bf2f(x1[cb + i1]);
        float v2 = bf2f(x1[cb + i2]);
        float y0 = A0 * v0 + B0, y1 = A1 * v1 + B1, y2 = A2 * v2 + B2;
        float t1 = y0 - y1, t2 = y0 - y2;
        t1s += t1 * t1;        t2s += t2 * t2;
        pa0 += (t1 - t2) * v0; pb0 += (t1 - t2);
        pa1 += -t1 * v1;       pb1 += -t1;
        pa2 += t2 * v2;        pb2 += t2;
    }

    // xdd over unfold(x) channels (Cin*9)
    float xdd = 0;
    int h0 = i0 / W, w0 = i0 - h0 * W;
    int h1 = i1 / W, w1 = i1 - h1 * W;
    int h2 = i2 / W, w2 = i2 - h2 * W;
    for (int cc = lane; cc < Cin * 9; cc += 32) {
        int ci = cc / 9, rem = cc - ci * 9;
        int kh = rem / 3 - 1, kw = rem - (rem / 3) * 3 - 1;
        size_t cb = ((size_t)(b * Cin + ci)) * HW;
        int a_h = h0 + kh, a_w = w0 + kw;
        int b_h = h1 + kh, b_w = w1 + kw;
        int c_h = h2 + kh, c_w = w2 + kw;
        float f0 = ((unsigned)a_h < (unsigned)H && (unsigned)a_w < (unsigned)W) ? bf2f(xin[cb + a_h * W + a_w]) : 0.0f;
        float f1 = ((unsigned)b_h < (unsigned)H && (unsigned)b_w < (unsigned)W) ? bf2f(xin[cb + b_h * W + b_w]) : 0.0f;
        float f2 = ((unsigned)c_h < (unsigned)H && (unsigned)c_w < (unsigned)W) ? bf2f(xin[cb + c_h * W + c_w]) : 0.0f;
        float d1 = f0 - f1, d2 = f0 - f2;
        xdd += d1 * d1 - d2 * d2;
    }

    // wave reductions (wave32)
    #pragma unroll
    for (int off = 16; off > 0; off >>= 1) {
        t1s += __shfl_xor(t1s, off); t2s += __shfl_xor(t2s, off);
        pa0 += __shfl_xor(pa0, off); pb0 += __shfl_xor(pb0, off);
        pa1 += __shfl_xor(pa1, off); pb1 += __shfl_xor(pb1, off);
        pa2 += __shfl_xor(pa2, off); pb2 += __shfl_xor(pb2, off);
        xdd += __shfl_xor(xdd, off);
    }

    if (lane == 0) {
        float invC  = 1.0f / (float)C;
        float x1dd  = (t1s - t2s) * invC;
        float sgn   = (xdd > 0.f) ? 1.f : ((xdd < 0.f) ? -1.f : 0.f);
        if (-sgn * x1dd > 0.f) {  // hinge active
            float coef = -sgn * 2.0f * invC * ((float)sample_num / (float)Bn);
            atomicAdd(&gA[base + i0], coef * pa0); atomicAdd(&gB[base + i0], coef * pb0);
            atomicAdd(&gA[base + i1], coef * pa1); atomicAdd(&gB[base + i1], coef * pb1);
            atomicAdd(&gA[base + i2], coef * pa2); atomicAdd(&gB[base + i2], coef * pb2);
        }
    }
}

// RMSprop step (adds dense gamma-regularizer gradient), zeroes grads for next step
__global__ void k_zzz_update(float* A, float* B, float* vA, float* vB,
                             float* gA, float* gB, int total, float reg_scale) {
    int i = blockIdx.x * blockDim.x + threadIdx.x;
    if (i >= total) return;
    float ga = gA[i] + reg_scale * (A[i] - 1.0f);
    float gb = gB[i] + reg_scale * B[i];
    float va = 0.99f * vA[i] + 0.01f * ga * ga;
    float vb = 0.99f * vB[i] + 0.01f * gb * gb;
    A[i] -= 0.001f * ga / (sqrtf(va) + 1e-8f);
    B[i] -= 0.001f * gb / (sqrtf(vb) + 1e-8f);
    vA[i] = va; vB[i] = vb;
    gA[i] = 0.0f; gB[i] = 0.0f;
}

// relu(bn(A*x1 + B)) -> bf16
__global__ void k_zzz_apply(const bf16_t* __restrict__ x1,
                            const float* __restrict__ A, const float* __restrict__ B,
                            const float* __restrict__ scale, const float* __restrict__ shift,
                            bf16_t* __restrict__ out, int C, int HW, int total) {
    int i = blockIdx.x * blockDim.x + threadIdx.x;
    if (i >= total) return;
    int p = i % HW;
    int t = i / HW;
    int c = t % C;
    int b = t / C;
    float v = A[(size_t)b * HW + p] * bf2f(x1[i]) + B[(size_t)b * HW + p];
    v = scale[c] * v + shift[c];
    out[i] = f2bf(v > 0.0f ? v : 0.0f);
}

// ---------------------------------------------------------------------------
__global__ void k_maxpool2(const bf16_t* __restrict__ in, bf16_t* __restrict__ out,
                           int C, int H, int W, int total) {
    int i = blockIdx.x * blockDim.x + threadIdx.x;
    if (i >= total) return;
    int Ho = H >> 1, Wo = W >> 1;
    int wo = i % Wo; int t = i / Wo;
    int ho = t % Ho; t /= Ho;
    int c  = t % C;  int b = t / C;
    size_t base = ((size_t)(b * C + c)) * H * W + (size_t)(ho * 2) * W + wo * 2;
    float m01 = fmaxf(bf2f(in[base]), bf2f(in[base + 1]));
    float m23 = fmaxf(bf2f(in[base + W]), bf2f(in[base + W + 1]));
    out[i] = f2bf(fmaxf(m01, m23));
}

__global__ void k_fc(const bf16_t* __restrict__ h, const float* __restrict__ wf,
                     const float* __restrict__ bfc, float* __restrict__ out, int Bn) {
    int i = blockIdx.x * blockDim.x + threadIdx.x;
    if (i >= Bn * 10) return;
    int b = i / 10, o = i - b * 10;
    float s = bfc[o];
    for (int k = 0; k < 512; ++k)
        s += bf2f(h[(size_t)b * 512 + k]) * wf[(size_t)o * 512 + k];
    out[i] = s;
}

// ---------------------------------------------------------------------------
// Host orchestration
// ---------------------------------------------------------------------------
struct LSpec { int Cin, Cout, H, samples, pool; };

extern "C" void kernel_launch(void* const* d_in, const int* in_sizes, int n_in,
                              void* d_out, int out_size, void* d_ws, size_t ws_size,
                              hipStream_t stream) {
    (void)in_sizes; (void)n_in; (void)out_size; (void)ws_size;
    const int BN = 128;

    const float* x = (const float*)d_in[0];
    auto Wp = [&](int L) { return (const float*)d_in[1 + (L - 1) * 6 + 0]; };
    auto Cp = [&](int L) { return (const float*)d_in[1 + (L - 1) * 6 + 1]; };
    auto Gp = [&](int L) { return (const float*)d_in[1 + (L - 1) * 6 + 2]; };
    auto Bp = [&](int L) { return (const float*)d_in[1 + (L - 1) * 6 + 3]; };
    auto Mp = [&](int L) { return (const float*)d_in[1 + (L - 1) * 6 + 4]; };
    auto Vp = [&](int L) { return (const float*)d_in[1 + (L - 1) * 6 + 5]; };
    const float* fcw = (const float*)d_in[1 + 13 * 6 + 0];
    const float* fcb = (const float*)d_in[1 + 13 * 6 + 1];

    // workspace carve (~60 MB)
    char* ws = (char*)d_ws;
    bf16_t* actA  = (bf16_t*)(ws + (size_t)0);
    bf16_t* actB  = (bf16_t*)(ws + ((size_t)16 << 20));
    bf16_t* x1buf = (bf16_t*)(ws + ((size_t)32 << 20));
    bf16_t* wbuf  = (bf16_t*)(ws + ((size_t)48 << 20));
    float*  scl   = (float*) (ws + ((size_t)56 << 20));
    float*  shf   = scl + 512;
    float*  Amap  = (float*) (ws + ((size_t)56 << 20) + 65536);
    float*  Bmap  = Amap + 131072;
    float*  vA    = Bmap + 131072;
    float*  vB    = vA   + 131072;
    float*  gA    = vB   + 131072;
    float*  gB    = gA   + 131072;

    // input fp32 -> bf16
    {
        int tot = BN * 3 * 32 * 32;
        k_f32_to_bf16<<<(tot + 255) / 256, 256, 0, stream>>>(x, actA, tot);
    }

    const LSpec specs[13] = {
        {3,   64, 32, 256, 0}, {64,  64, 32, 0, 1},
        {64, 128, 16, 128, 0}, {128,128, 16, 0, 1},
        {128,256,  8,  64, 0}, {256,256,  8, 0, 0}, {256,256, 8, 0, 1},
        {256,512,  4,  64, 0}, {512,512,  4, 0, 0}, {512,512, 4, 0, 1},
        {512,512,  2,   0, 0}, {512,512,  2, 0, 0}, {512,512, 2, 0, 1}
    };

    bf16_t* cur = actA;
    bf16_t* nxt = actB;

    for (int L = 1; L <= 13; ++L) {
        const LSpec sp = specs[L - 1];
        const int H = sp.H, Wd = sp.H, HW = H * Wd;
        const int K = sp.Cin * 9, Kpad = (K + 31) & ~31;
        const int Ntot = BN * HW;
        const bool zzz = sp.samples > 0;

        {   // pack weights, fold BN
            int wtot = sp.Cout * Kpad;
            k_pack_weights<<<(wtot + 255) / 256, 256, 0, stream>>>(Wp(L), wbuf, sp.Cout, K, Kpad);
            k_scale_shift<<<(sp.Cout + 63) / 64, 64, 0, stream>>>(
                Gp(L), Bp(L), Mp(L), Vp(L), Cp(L), scl, shf, sp.Cout, zzz ? 0 : 1);
        }

        dim3 grid(Ntot / 128, sp.Cout / 64);
        if (!zzz) {
            k_conv_wmma<<<grid, 256, 0, stream>>>(cur, wbuf, scl, shf, Cp(L), nxt,
                                                  sp.Cin, sp.Cout, H, Wd, K, Kpad, 0);
        } else {
            k_conv_wmma<<<grid, 256, 0, stream>>>(cur, wbuf, scl, shf, Cp(L), x1buf,
                                                  sp.Cin, sp.Cout, H, Wd, K, Kpad, 1);
            const int ptot = BN * HW;
            k_zzz_init<<<(ptot + 255) / 256, 256, 0, stream>>>(Amap, Bmap, vA, vB, gA, gB, ptot);
            const float reg_scale = 1000.0f * 2.0f / (float)ptot;
            const int nsamp = BN * sp.samples;
            for (int st = 0; st < 10; ++st) {
                k_zzz_grad<<<(nsamp + 7) / 8, 256, 0, stream>>>(
                    cur, x1buf, Amap, Bmap, gA, gB,
                    sp.Cin, sp.Cout, H, Wd, BN, sp.samples, st, L);
                k_zzz_update<<<(ptot + 255) / 256, 256, 0, stream>>>(
                    Amap, Bmap, vA, vB, gA, gB, ptot, reg_scale);
            }
            const int atot = BN * sp.Cout * HW;
            k_zzz_apply<<<(atot + 255) / 256, 256, 0, stream>>>(
                x1buf, Amap, Bmap, scl, shf, nxt, sp.Cout, HW, atot);
        }
        { bf16_t* t = cur; cur = nxt; nxt = t; }

        if (sp.pool) {
            const int ptot = BN * sp.Cout * (H / 2) * (Wd / 2);
            k_maxpool2<<<(ptot + 255) / 256, 256, 0, stream>>>(cur, nxt, sp.Cout, H, Wd, ptot);
            bf16_t* t = cur; cur = nxt; nxt = t;
        }
    }

    k_fc<<<(BN * 10 + 255) / 256, 256, 0, stream>>>(cur, fcw, fcb, (float*)d_out, BN);
}
